// Decoder_69595650064745
// MI455X (gfx1250) — compile-verified, matrix operand order checked by reading, not run on hardware
//
#include <hip/hip_runtime.h>

// ---------------- problem constants ----------------
#define B_   64
#define T_   512
#define I_   512
#define H1_  256
#define H2_  128
#define C_   1024
#define M_   80
#define NWG  32          // persistent workgroups (one 32-channel slice each)
#define NTHR 256         // 8 waves of 32
#define KC   256         // K-chunk staged in LDS (64*256*2B = 32KB)

typedef __bf16 bhalf;
typedef bhalf v16bf __attribute__((ext_vector_type(16)));
typedef float v8f   __attribute__((ext_vector_type(8)));
typedef unsigned int v4u __attribute__((ext_vector_type(4)));
typedef int v4i __attribute__((ext_vector_type(4)));

#define AS1 __attribute__((address_space(1)))
#define AS3 __attribute__((address_space(3)))

#if __has_builtin(__builtin_amdgcn_global_load_async_to_lds_b128) && \
    __has_builtin(__builtin_amdgcn_s_wait_asynccnt)
#define HAVE_ASYNC_LDS 1
#else
#define HAVE_ASYNC_LDS 0
#endif

// ---------------- workspace layout (bytes) ----------------
constexpr size_t SZ_AIH = (size_t)3 * C_ * (I_ + H2_);   // 3072*640
constexpr size_t SZ_HH  = (size_t)3 * C_ * C_;           // 3072*1024
constexpr size_t OFF_W_AIH = 0;
constexpr size_t OFF_W_AHH = OFF_W_AIH + SZ_AIH * 2;
constexpr size_t OFF_W_GIH = OFF_W_AHH + SZ_HH * 2;      // 2 layers contiguous
constexpr size_t OFF_W_GHH = OFF_W_GIH + 2 * SZ_HH * 2;
constexpr size_t OFF_W_PRJ = OFF_W_GHH + 2 * SZ_HH * 2;
constexpr size_t OFF_FRAME = OFF_W_PRJ + (size_t)M_ * C_ * 2;   // f32 [64*80]
constexpr size_t OFF_PRE1  = OFF_FRAME + (size_t)B_ * M_ * 4;   // f32 [64*256]
constexpr size_t OFF_ATTNH = OFF_PRE1  + (size_t)B_ * H1_ * 4;  // f32 [64*1024]
constexpr size_t OFF_H1    = OFF_ATTNH + (size_t)B_ * C_ * 4;
constexpr size_t OFF_H2    = OFF_H1    + (size_t)B_ * C_ * 4;
constexpr size_t OFF_X2    = OFF_H2    + (size_t)B_ * C_ * 4;
constexpr size_t OFF_XOUT  = OFF_X2    + (size_t)B_ * C_ * 4;
constexpr size_t OFF_XCATB = OFF_XOUT  + (size_t)B_ * C_ * 4;           // bf16 [64*640]
constexpr size_t OFF_AHBF  = OFF_XCATB + (size_t)B_ * (I_ + H2_) * 2;   // bf16 x2 (ping-pong)
constexpr size_t OFF_H1BF  = OFF_AHBF  + 2 * (size_t)B_ * C_ * 2;
constexpr size_t OFF_H2BF  = OFF_H1BF  + 2 * (size_t)B_ * C_ * 2;
constexpr size_t OFF_X2B   = OFF_H2BF  + 2 * (size_t)B_ * C_ * 2;       // bf16
constexpr size_t OFF_XOB   = OFF_X2B   + (size_t)B_ * C_ * 2;           // bf16
constexpr size_t OFF_BAR   = OFF_XOB   + (size_t)B_ * C_ * 2;           // 2 counters (padded)
constexpr size_t WS_END    = OFF_BAR + 256;

// ---------------- address-space helpers (force global_/ds_ ops, not flat_) ----
template <typename T>
__device__ __forceinline__ AS1 const T* gcp(const T* p) {
  return (AS1 const T*)(unsigned long long)p;
}
template <typename T>
__device__ __forceinline__ AS1 T* gp(T* p) {
  return (AS1 T*)(unsigned long long)p;
}

template <typename T>
__device__ __forceinline__ T gld(const T* p) { return *gcp(p); }
template <typename T>
__device__ __forceinline__ void gst(T* p, T v) { *gp(p) = v; }

// ---------------- small helpers ----------------
__device__ __forceinline__ unsigned short f2bf(float f) {
  unsigned u = __float_as_uint(f);
  u += 0x7fffu + ((u >> 16) & 1u);          // round-to-nearest-even
  return (unsigned short)(u >> 16);
}
__device__ __forceinline__ float sigm(float x) { return 1.f / (1.f + __expf(-x)); }
__device__ __forceinline__ float tanh_f(float x) { float e = __expf(-2.f * x); return (1.f - e) / (1.f + e); }

// 16-bit operand fragment (ISA 7.12.2): lane l -> row (l&15), k-group (l>>4)*8,
// elements 0..7 at k, elements 8..15 at k+16.  p points at row_base + kgroup.
__device__ __forceinline__ v16bf ldfrag_g(const unsigned short* p) {
  union { v16bf v; v4u u[2]; } x;
  x.u[0] = *(AS1 const v4u*)(unsigned long long)p;
  x.u[1] = *(AS1 const v4u*)(unsigned long long)(p + 16);
  return x.v;
}
__device__ __forceinline__ v16bf ldfrag_s(const unsigned short* p) {
  union { v16bf v; v4u u[2]; } x;
  x.u[0] = *(AS3 const v4u*)(unsigned)(unsigned long long)p;
  x.u[1] = *(AS3 const v4u*)(unsigned)(unsigned long long)(p + 16);
  return x.v;
}

__device__ __forceinline__ v8f wmma_bf16(v16bf a, v16bf b, v8f c) {
  return __builtin_amdgcn_wmma_f32_16x16x32_bf16(false, a, false, b, (short)0, c,
                                                 false, false);
}

// grid-wide barrier (persistent kernel, exactly NWG resident blocks)
__device__ __forceinline__ void gbar(unsigned* cnt, volatile unsigned* gen) {
  __syncthreads();
  if (threadIdx.x == 0) {
    __threadfence();
    unsigned g = *gen;
    if (atomicAdd(cnt, 1u) == NWG - 1u) {
      *cnt = 0u;
      __threadfence();
      atomicAdd((unsigned*)gen, 1u);
    } else {
      while (*gen == g) { __builtin_amdgcn_s_sleep(2); }
    }
    __threadfence();
  }
  __syncthreads();
}

// 3-gate GEMM pass: acc{0,1,2} += A[64xK] * W[wr{0,1,2}][K]^T tile for this wave.
// A staged through LDS in KC-chunks (async DMA when available);
// W (bf16, row-major [3C x K]) streamed from L2 as global_load_b128.
__device__ void mm3_pass(v8f& c0, v8f& c1, v8f& c2,
                         const unsigned short* Abf, int K,
                         const unsigned short* W, int wr0, int wr1, int wr2,
                         unsigned short* As, int tid, int am, int g8) {
  for (int kb = 0; kb < K; kb += KC) {
    int cur = (K - kb < KC) ? (K - kb) : KC;
    __syncthreads();                           // previous chunk fully consumed
    int nvec = (64 * cur) >> 3;                // 8 bf16 per b128; divides NTHR exactly
#if HAVE_ASYNC_LDS
    for (int i = tid; i < nvec; i += NTHR) {
      int e = i << 3;
      int r = e / cur, cc = e % cur;
      __builtin_amdgcn_global_load_async_to_lds_b128(
          (AS1 v4i*)(unsigned long long)&Abf[(size_t)r * K + kb + cc],
          (AS3 v4i*)(unsigned)(unsigned long long)&As[r * KC + cc],
          0, 0);
    }
    __builtin_amdgcn_s_wait_asynccnt(0);
#else
    for (int i = tid; i < nvec; i += NTHR) {
      int e = i << 3;
      int r = e / cur, cc = e % cur;
      *(AS3 v4u*)(unsigned)(unsigned long long)&As[r * KC + cc] =
          *(AS1 const v4u*)(unsigned long long)&Abf[(size_t)r * K + kb + cc];
    }
#endif
    __syncthreads();
    const unsigned short* ap = As + am * KC + g8;
    const unsigned short* b0 = W + (size_t)wr0 * K + kb + g8;
    const unsigned short* b1 = W + (size_t)wr1 * K + kb + g8;
    const unsigned short* b2 = W + (size_t)wr2 * K + kb + g8;
    for (int k = 0; k < cur; k += 32) {
      v16bf a  = ldfrag_s(ap + k);
      v16bf f0 = ldfrag_g(b0 + k);
      v16bf f1 = ldfrag_g(b1 + k);
      v16bf f2 = ldfrag_g(b2 + k);
      c0 = wmma_bf16(a, f0, c0);
      c1 = wmma_bf16(a, f1, c1);
      c2 = wmma_bf16(a, f2, c2);
    }
  }
}

// One GRU cell stage for this WG's 32-channel slice.
__device__ void gru_stage(const unsigned short* xbf, int K,
                          const unsigned short* wih, const unsigned short* whh,
                          const unsigned short* hbf_rd, unsigned short* hbf_wr,
                          const float* bih, const float* bhh,
                          float* h_state,
                          const float* xres, float* xout, unsigned short* xout_bf,
                          unsigned short* As, int tid) {
  int wv = tid >> 5, lane = tid & 31;
  int li = lane & 15, g8 = (lane >> 4) * 8;
  int rt = wv & 3, ct = wv >> 2;
  int am = rt * 16 + li;                    // batch row for A fragment
  int nch = blockIdx.x * 32 + ct * 16 + li; // output channel for B fragment

  v8f ir = {}, iz = {}, in_ = {}, hr = {}, hz = {}, hn = {};
  mm3_pass(ir, iz, in_, xbf, K, wih, nch, C_ + nch, 2 * C_ + nch, As, tid, am, g8);
  mm3_pass(hr, hz, hn, hbf_rd, C_, whh, nch, C_ + nch, 2 * C_ + nch, As, tid, am, g8);

  float bri = gld(bih + nch), bzi = gld(bih + C_ + nch), bni = gld(bih + 2 * C_ + nch);
  float brh = gld(bhh + nch), bzh = gld(bhh + C_ + nch), bnh = gld(bhh + 2 * C_ + nch);
#pragma unroll
  for (int i = 0; i < 8; ++i) {
    int b = rt * 16 + i + g8;               // C/D layout: VGPR i -> M=i (+8 hi half)
    float r = sigm(ir[i] + bri + hr[i] + brh);
    float z = sigm(iz[i] + bzi + hz[i] + bzh);
    float n = tanh_f(in_[i] + bni + r * (hn[i] + bnh));
    float hold = gld(h_state + (size_t)b * C_ + nch);
    float hnew = (1.f - z) * n + z * hold;
    gst(h_state + (size_t)b * C_ + nch, hnew);
    gst(hbf_wr + (size_t)b * C_ + nch, f2bf(hnew));
    if (xres) {
      float xv = gld(xres + (size_t)b * C_ + nch) + hnew;
      gst(xout + (size_t)b * C_ + nch, xv);
      gst(xout_bf + (size_t)b * C_ + nch, f2bf(xv));
    }
  }
}

// ---------------- prep kernels ----------------
__global__ void cvt_bf(const float* __restrict__ s, unsigned short* __restrict__ d, size_t n) {
  size_t i = (size_t)blockIdx.x * blockDim.x + threadIdx.x;
  size_t st = (size_t)gridDim.x * blockDim.x;
  for (; i < n; i += st) gst(d + i, f2bf(gld(s + i)));
}
__global__ void zero_ws(unsigned* __restrict__ p, size_t n) {
  size_t i = (size_t)blockIdx.x * blockDim.x + threadIdx.x;
  size_t st = (size_t)gridDim.x * blockDim.x;
  for (; i < n; i += st) gst(p + i, 0u);
}

// ---------------- persistent decoder ----------------
extern "C" __global__ __launch_bounds__(NTHR)
void decoder_kernel(const float* __restrict__ inputs,
                    const float* __restrict__ pre_w0, const float* __restrict__ pre_b0,
                    const float* __restrict__ pre_w1, const float* __restrict__ pre_b1,
                    const float* __restrict__ attn_b_ih, const float* __restrict__ attn_b_hh,
                    const float* __restrict__ gru_b_ih, const float* __restrict__ gru_b_hh,
                    const float* __restrict__ proj_b,
                    float* __restrict__ out, char* ws) {
  __shared__ __attribute__((aligned(16))) unsigned short As[64 * KC];  // 32 KB

  const unsigned short* w_aih = (const unsigned short*)(ws + OFF_W_AIH);
  const unsigned short* w_ahh = (const unsigned short*)(ws + OFF_W_AHH);
  const unsigned short* w_gih = (const unsigned short*)(ws + OFF_W_GIH);
  const unsigned short* w_ghh = (const unsigned short*)(ws + OFF_W_GHH);
  const unsigned short* w_prj = (const unsigned short*)(ws + OFF_W_PRJ);
  float* frame  = (float*)(ws + OFF_FRAME);
  float* pre1   = (float*)(ws + OFF_PRE1);
  float* attnh  = (float*)(ws + OFF_ATTNH);
  float* h1     = (float*)(ws + OFF_H1);
  float* h2     = (float*)(ws + OFF_H2);
  float* x2f    = (float*)(ws + OFF_X2);
  float* xoutf  = (float*)(ws + OFF_XOUT);
  unsigned short* xcat = (unsigned short*)(ws + OFF_XCATB);
  unsigned short* ahbf = (unsigned short*)(ws + OFF_AHBF);   // [2][64*1024]
  unsigned short* h1bf = (unsigned short*)(ws + OFF_H1BF);
  unsigned short* h2bf = (unsigned short*)(ws + OFF_H2BF);
  unsigned short* x2b  = (unsigned short*)(ws + OFF_X2B);
  unsigned short* xob  = (unsigned short*)(ws + OFF_XOB);
  unsigned* bcnt = (unsigned*)(ws + OFF_BAR);
  volatile unsigned* bgen = (volatile unsigned*)(ws + OFF_BAR + 128);

  const int tid = threadIdx.x;
  const int gtid = blockIdx.x * NTHR + tid;
  const size_t HB = (size_t)B_ * C_;

  for (int t = 0; t <= T_; ++t) {
    int par = t & 1;

    // ---- S0: projection of x_out(t-1) -> frame(t-1); write output frame ----
    if (t > 0) {
      int w = blockIdx.x * 8 + (tid >> 5);
      if (w < 20) {                          // 4 row-tiles x 5 col-tiles of [64x80]
        int lane = tid & 31, li = lane & 15, g8 = (lane >> 4) * 8;
        int rt = w & 3, ctile = w >> 2;
        const unsigned short* ap = xob + (size_t)(rt * 16 + li) * C_ + g8;
        const unsigned short* bp = w_prj + (size_t)(ctile * 16 + li) * C_ + g8;
        v8f acc = {};
        for (int k = 0; k < C_; k += 32)
          acc = wmma_bf16(ldfrag_g(ap + k), ldfrag_g(bp + k), acc);
        int mcol = ctile * 16 + li;
#pragma unroll
        for (int i = 0; i < 8; ++i) {
          int b = rt * 16 + i + g8;
          float v = acc[i] + gld(proj_b + mcol);
          gst(out + ((size_t)b * T_ + (t - 1)) * M_ + mcol, v);
          gst(frame + (size_t)b * M_ + mcol, v);
        }
      }
    }
    gbar(bcnt, bgen);
    if (t == T_) break;

    // ---- P0: prenet layer 0 (fp32 VALU; tiny) ----
    for (int idx = gtid; idx < B_ * H1_; idx += NWG * NTHR) {
      int b = idx >> 8, j = idx & 255;
      float acc = gld(pre_b0 + j);
      const float* fr = frame + (size_t)b * M_;
      const float* wr = pre_w0 + (size_t)j * M_;
#pragma unroll 4
      for (int k = 0; k < M_; ++k) acc += gld(fr + k) * gld(wr + k);
      gst(pre1 + idx, acc > 0.f ? acc : 0.f);
    }
    gbar(bcnt, bgen);

    // ---- P1: prenet layer 1 -> xcat[:,512:640]; convert feat_t -> xcat[:,0:512] ----
    {
      int b = gtid >> 7, j = gtid & 127;     // 8192 threads == B_*H2_
      float acc = gld(pre_b1 + j);
      const float* pr = pre1 + (size_t)b * H1_;
      const float* wr = pre_w1 + (size_t)j * H1_;
#pragma unroll 4
      for (int k = 0; k < H1_; ++k) acc += gld(pr + k) * gld(wr + k);
      gst(xcat + (size_t)b * 640 + 512 + j, f2bf(acc > 0.f ? acc : 0.f));
#pragma unroll
      for (int e = 0; e < 4; ++e) {
        int fi = gtid + e * (NWG * NTHR);    // 0..32767
        int fb = fi >> 9, fk = fi & 511;
        gst(xcat + (size_t)fb * 640 + fk,
            f2bf(gld(inputs + ((size_t)fb * T_ + t) * I_ + fk)));
      }
    }
    gbar(bcnt, bgen);

    // ---- S1: attention GRU ----
    gru_stage(xcat, I_ + H2_, w_aih, w_ahh,
              ahbf + (size_t)par * HB, ahbf + (size_t)(par ^ 1) * HB,
              attn_b_ih, attn_b_hh, attnh,
              nullptr, nullptr, nullptr, As, tid);
    gbar(bcnt, bgen);

    // ---- S2: residual GRU layer 0 ----
    gru_stage(ahbf + (size_t)(par ^ 1) * HB, C_, w_gih, w_ghh,
              h1bf + (size_t)par * HB, h1bf + (size_t)(par ^ 1) * HB,
              gru_b_ih, gru_b_hh, h1,
              attnh, x2f, x2b, As, tid);
    gbar(bcnt, bgen);

    // ---- S3: residual GRU layer 1 ----
    gru_stage(x2b, C_, w_gih + SZ_HH, w_ghh + SZ_HH,
              h2bf + (size_t)par * HB, h2bf + (size_t)(par ^ 1) * HB,
              gru_b_ih + 3 * C_, gru_b_hh + 3 * C_, h2,
              x2f, xoutf, xob, As, tid);
    gbar(bcnt, bgen);
  }
}

// ---------------- host ----------------
extern "C" void kernel_launch(void* const* d_in, const int* in_sizes, int n_in,
                              void* d_out, int out_size, void* d_ws, size_t ws_size,
                              hipStream_t stream) {
  (void)in_sizes; (void)n_in; (void)out_size; (void)ws_size;
  char* ws = (char*)d_ws;

  const float* inputs    = (const float*)d_in[0];
  const float* pre_w0    = (const float*)d_in[1];
  const float* pre_b0    = (const float*)d_in[2];
  const float* pre_w1    = (const float*)d_in[3];
  const float* pre_b1    = (const float*)d_in[4];
  const float* attn_w_ih = (const float*)d_in[5];
  const float* attn_w_hh = (const float*)d_in[6];
  const float* attn_b_ih = (const float*)d_in[7];
  const float* attn_b_hh = (const float*)d_in[8];
  const float* gru_w_ih  = (const float*)d_in[9];
  const float* gru_w_hh  = (const float*)d_in[10];
  const float* gru_b_ih  = (const float*)d_in[11];
  const float* gru_b_hh  = (const float*)d_in[12];
  const float* proj_w    = (const float*)d_in[13];
  const float* proj_b    = (const float*)d_in[14];

  // fp32 -> bf16 weight conversion (L2-resident working set for decode loop)
  cvt_bf<<<512, 256, 0, stream>>>(attn_w_ih, (unsigned short*)(ws + OFF_W_AIH), SZ_AIH);
  cvt_bf<<<512, 256, 0, stream>>>(attn_w_hh, (unsigned short*)(ws + OFF_W_AHH), SZ_HH);
  cvt_bf<<<1024, 256, 0, stream>>>(gru_w_ih, (unsigned short*)(ws + OFF_W_GIH), 2 * SZ_HH);
  cvt_bf<<<1024, 256, 0, stream>>>(gru_w_hh, (unsigned short*)(ws + OFF_W_GHH), 2 * SZ_HH);
  cvt_bf<<<64, 256, 0, stream>>>(proj_w, (unsigned short*)(ws + OFF_W_PRJ), (size_t)M_ * C_);
  // zero all state + barrier counters (deterministic across replays)
  zero_ws<<<512, 256, 0, stream>>>((unsigned*)(ws + OFF_FRAME), (WS_END - OFF_FRAME) / 4);

  decoder_kernel<<<NWG, NTHR, 0, stream>>>(inputs, pre_w0, pre_b0, pre_w1, pre_b1,
                                           attn_b_ih, attn_b_hh, gru_b_ih, gru_b_hh,
                                           proj_b, (float*)d_out, ws);
}
